// GAT_53094385713560
// MI455X (gfx1250) — matrix operand order repaired and checked by reading, not compiled, not run on hardware
//
#include <hip/hip_runtime.h>
#include <math.h>

// ---------------- problem constants (match reference) ----------------
#define NN      100000
#define EE      1600000
#define ETOT    (EE + NN)       // edges + self loops = 1,700,000
#define GG      64
#define INDIM   128
#define HID     64
#define HEADS   4
#define CH      16              // HID / HEADS
#define OUTD    40
#define CATD    (HID * 3)       // 192
#define ATT_SLOPE 0.2f
#define ACT_SLOPE 0.01f
#define BN_RSQ  0.99999500003749969f   // 1/sqrt(1+1e-5)

typedef _Float16 v16h __attribute__((ext_vector_type(16)));
typedef float    v8f  __attribute__((ext_vector_type(8)));

// ---------------- helpers ----------------
__device__ __forceinline__ unsigned pack2h(float a, float b) {
    _Float16 h[2] = {(_Float16)a, (_Float16)b};
    unsigned u; __builtin_memcpy(&u, h, 4); return u;
}
__device__ __forceinline__ void atomicMaxF32(float* addr, float val) {
    // sign-split trick: int-max for >=0, uint-min for <0; init with -inf bits
    if (val >= 0.f) atomicMax((int*)addr, __float_as_int(val));
    else            atomicMin((unsigned int*)addr, __float_as_uint(val));
}

// ---------------- fill ----------------
__global__ void fill_f32(float* __restrict__ p, float v, int n) {
    int i = blockIdx.x * 256 + threadIdx.x;
    if (i < n) p[i] = v;
}

// ---------------- WMMA GEMM:  Y[nRows,64] = X[nRows,K](ldx) @ W[K,64]  ----------------
// act==0: raw store (bias ignored). act==1: BN(lrelu(acc+bias)) with gamma/beta.
//
// LDS holds W pre-converted to f16 in *B-fragment order*: for kstep ks, K-half kh,
// column col, the 8 packed half2 dwords of that lane's fragment are contiguous, so
// each B fragment is exactly two ds_load_b128s into consecutive VGPRs.
//   ldsB[((ks*2 + kh) << 9) + (col << 3) + j]  = half2( W[kg][col], W[kg+1][col] ),
//   kg = ks*32 + kh*8 + 2j + (j >= 4 ? 8 : 0)
__global__ __launch_bounds__(128) void gemm_wmma_kernel(
    const float* __restrict__ X, int ldx,
    const float* __restrict__ Wm,
    const float* __restrict__ bias,
    const float* __restrict__ gam,
    const float* __restrict__ beta,
    float* __restrict__ Y, int ldy,
    int nRows, int K, int act)
{
    __shared__ __align__(16) unsigned ldsB[6 * 1024];   // supports K<=192 (6 ksteps)

    const int tid = threadIdx.x;
    const int npack = K * 32;                           // (K/2)*64 packed dwords
    for (int q = tid; q < npack; q += 128) {
        const int ks  = q >> 10;
        const int kh  = (q >> 9) & 1;
        const int col = (q >> 3) & 63;
        const int j   = q & 7;
        const int kg  = ks * 32 + kh * 8 + 2 * j + ((j >> 2) << 3);
        ldsB[q] = pack2h(Wm[kg * 64 + col], Wm[(kg + 1) * 64 + col]);
    }
    __syncthreads();

    const int wave  = tid >> 5;
    const int lane  = tid & 31;
    const int rlane = lane & 15;                   // row-in-tile / col-in-tile
    const int khalf = lane >> 4;                   // K-half selector
    const int tileRow = blockIdx.x * 64 + wave * 16;
    if (tileRow >= nRows) return;                  // whole-wave exit: EXEC stays all-1s

    v8f acc[4];
#pragma unroll
    for (int t = 0; t < 4; ++t) acc[t] = (v8f){0,0,0,0,0,0,0,0};

    const float* xr = X + (size_t)(tileRow + rlane) * ldx;

    for (int kb = 0; kb < K; kb += 32) {
        if (kb + 32 < K) __builtin_prefetch(xr + kb + 32, 0, 3);   // WGP-scope prefetch

        // A fragment, 16-bit 16x32 layout: lanes 0-15 -> K {0..7,16..23}, lanes 16-31 -> +8
        v16h a;
        const float4 f0 = *(const float4*)(xr + kb + khalf * 8);
        const float4 f1 = *(const float4*)(xr + kb + khalf * 8 + 4);
        const float4 f2 = *(const float4*)(xr + kb + 16 + khalf * 8);
        const float4 f3 = *(const float4*)(xr + kb + 16 + khalf * 8 + 4);
        a[0]=(_Float16)f0.x;  a[1]=(_Float16)f0.y;  a[2]=(_Float16)f0.z;  a[3]=(_Float16)f0.w;
        a[4]=(_Float16)f1.x;  a[5]=(_Float16)f1.y;  a[6]=(_Float16)f1.z;  a[7]=(_Float16)f1.w;
        a[8]=(_Float16)f2.x;  a[9]=(_Float16)f2.y;  a[10]=(_Float16)f2.z; a[11]=(_Float16)f2.w;
        a[12]=(_Float16)f3.x; a[13]=(_Float16)f3.y; a[14]=(_Float16)f3.z; a[15]=(_Float16)f3.w;

        const int fragBase = (((kb >> 5) * 2 + khalf) << 9) + (rlane << 3);
#pragma unroll
        for (int t = 0; t < 4; ++t) {
            // two contiguous 128-bit LDS loads -> full B fragment in consecutive VGPRs
            const uint4* bp = (const uint4*)&ldsB[fragBase + (t << 7)];
            const uint4 b0 = bp[0];
            const uint4 b1 = bp[1];
            v16h b;
            __builtin_memcpy(&b, &b0, 16);
            __builtin_memcpy(((char*)&b) + 16, &b1, 16);
            acc[t] = __builtin_amdgcn_wmma_f32_16x16x32_f16(
                false, a, false, b, (short)0, acc[t], false, false);
        }
    }

    // C/D layout: lane col = rlane, rows r + khalf*8
#pragma unroll
    for (int t = 0; t < 4; ++t) {
#pragma unroll
        for (int r = 0; r < 8; ++r) {
            const int orow = tileRow + r + khalf * 8;
            const int ocol = t * 16 + rlane;
            float v = acc[t][r];
            if (act) {
                v += bias[ocol];
                v = v >= 0.f ? v : ACT_SLOPE * v;
                v = gam[ocol] * (v * BN_RSQ) + beta[ocol];
            }
            Y[(size_t)orow * ldy + ocol] = v;
        }
    }
}

// ---------------- per-node attention logits ----------------
__global__ void att_nodes_kernel(const float* __restrict__ H,
                                 const float* __restrict__ att_src,
                                 const float* __restrict__ att_dst,
                                 float* __restrict__ a_src,
                                 float* __restrict__ a_dst)
{
    int i = blockIdx.x * 256 + threadIdx.x;
    if (i >= NN * HEADS) return;
    int node = i >> 2, h = i & 3;
    const float* hp = H + (size_t)node * HID + h * CH;
    float s = 0.f, d = 0.f;
#pragma unroll
    for (int c = 0; c < CH; ++c) {
        float v = hp[c];
        s += v * att_src[h * CH + c];
        d += v * att_dst[h * CH + c];
    }
    a_src[i] = s; a_dst[i] = d;
}

// ---------------- edge pass 1: alpha + segment max ----------------
__global__ void edge_alpha_max_kernel(const int* __restrict__ e0, const int* __restrict__ e1,
                                      const float* __restrict__ a_src, const float* __restrict__ a_dst,
                                      float* __restrict__ alpha, float* __restrict__ m)
{
    int i = blockIdx.x * 256 + threadIdx.x;
    if (i >= ETOT * HEADS) return;
    int e = i >> 2, h = i & 3;
    int s = (e < EE) ? e0[e] : (e - EE);
    int d = (e < EE) ? e1[e] : (e - EE);
    float a = a_src[s * HEADS + h] + a_dst[d * HEADS + h];
    a = a >= 0.f ? a : ATT_SLOPE * a;
    alpha[i] = a;
    atomicMaxF32(&m[d * HEADS + h], a);
}

// ---------------- edge pass 2: exp + segment sum ----------------
__global__ void edge_exp_sum_kernel(const int* __restrict__ e1,
                                    float* __restrict__ alpha,
                                    const float* __restrict__ m,
                                    float* __restrict__ denom)
{
    int i = blockIdx.x * 256 + threadIdx.x;
    if (i >= ETOT * HEADS) return;
    int e = i >> 2, h = i & 3;
    int d = (e < EE) ? e1[e] : (e - EE);
    float ex = __expf(alpha[i] - m[d * HEADS + h]);
    alpha[i] = ex;
    atomicAdd(&denom[d * HEADS + h], ex);
}

// ---------------- edge pass 3: weighted message aggregation ----------------
__global__ void edge_aggregate_kernel(const int* __restrict__ e0, const int* __restrict__ e1,
                                      const float* __restrict__ ex,
                                      const float* __restrict__ denom,
                                      const float* __restrict__ H,
                                      float* __restrict__ Hcat, int layerOff)
{
    int i = blockIdx.x * 256 + threadIdx.x;
    if (i >= ETOT * HEADS) return;
    int e = i >> 2, h = i & 3;
    int s = (e < EE) ? e0[e] : (e - EE);
    int d = (e < EE) ? e1[e] : (e - EE);
    float att = ex[i] / (denom[d * HEADS + h] + 1e-16f);
    const float* hs = H + (size_t)s * HID + h * CH;
    float* op = Hcat + (size_t)d * CATD + layerOff + h * CH;
#pragma unroll
    for (int c = 0; c < CH; ++c) atomicAdd(&op[c], hs[c] * att);
}

// ---------------- GAT epilogue: bias + ELU into concat slot ----------------
__global__ void bias_elu_kernel(float* __restrict__ Hcat, const float* __restrict__ bias, int layerOff)
{
    int i = blockIdx.x * 256 + threadIdx.x;
    if (i >= NN * HID) return;
    int node = i >> 6, c = i & 63;
    float* p = Hcat + (size_t)node * CATD + layerOff + c;
    float v = *p + bias[c];
    *p = v > 0.f ? v : (__expf(v) - 1.f);
}

// ---------------- mean pool ----------------
__global__ void pool_sum_kernel(const float* __restrict__ P, const int* __restrict__ batch,
                                float* __restrict__ sums)
{
    int i = blockIdx.x * 256 + threadIdx.x;
    if (i >= NN * HID) return;
    int node = i >> 6, c = i & 63;
    atomicAdd(&sums[batch[node] * HID + c], P[i]);
}
__global__ void pool_cnt_kernel(const int* __restrict__ batch, float* __restrict__ cnt)
{
    int i = blockIdx.x * 256 + threadIdx.x;
    if (i < NN) atomicAdd(&cnt[batch[i]], 1.f);
}
__global__ void pool_div_kernel(const float* __restrict__ sums, const float* __restrict__ cnt,
                                float* __restrict__ g)
{
    int i = blockIdx.x * 256 + threadIdx.x;
    if (i >= GG * HID) return;
    g[i] = sums[i] / fmaxf(cnt[i >> 6], 1.f);
}

// ---------------- tiny head linear (64 rows) ----------------
__global__ void head_linear_kernel(const float* __restrict__ X,
                                   const float* __restrict__ W,
                                   const float* __restrict__ b,
                                   const float* __restrict__ gam,
                                   const float* __restrict__ beta,
                                   float* __restrict__ Y,
                                   int rows, int K, int ncol, int act)
{
    int i = blockIdx.x * 256 + threadIdx.x;
    if (i >= rows * ncol) return;
    int r = i / ncol, c = i - r * ncol;
    float acc = b[c];
    for (int k = 0; k < K; ++k) acc += X[r * K + k] * W[k * ncol + c];
    if (act) {
        acc = acc >= 0.f ? acc : ACT_SLOPE * acc;
        acc = gam[c] * (acc * BN_RSQ) + beta[c];
    }
    Y[i] = acc;
}

// ---------------- log softmax over 40 cols ----------------
__global__ void logsoftmax_kernel(const float* __restrict__ L, float* __restrict__ out)
{
    __shared__ float sh[OUTD];
    __shared__ float mx, lse;
    int r = blockIdx.x, t = threadIdx.x;
    if (t < OUTD) sh[t] = L[r * OUTD + t];
    __syncthreads();
    if (t == 0) {
        float m = sh[0];
        for (int j = 1; j < OUTD; ++j) m = fmaxf(m, sh[j]);
        float s = 0.f;
        for (int j = 0; j < OUTD; ++j) s += __expf(sh[j] - m);
        mx = m; lse = __logf(s);
    }
    __syncthreads();
    if (t < OUTD) out[r * OUTD + t] = sh[t] - mx - lse;
}

// ---------------- host orchestration ----------------
static inline dim3 gb(int n) { return dim3((n + 255) / 256); }

extern "C" void kernel_launch(void* const* d_in, const int* in_sizes, int n_in,
                              void* d_out, int out_size, void* d_ws, size_t ws_size,
                              hipStream_t stream)
{
    (void)in_sizes; (void)n_in; (void)out_size; (void)ws_size;

    // -------- inputs (setup_inputs dict order, nested structures in insertion order) --------
    const float* x    = (const float*)d_in[0];            // [N,128]
    const int*   ei   = (const int*)d_in[1];              // [2,E] (jax default int32)
    const int*   e0   = ei;
    const int*   e1   = ei + EE;
    const int*   batc = (const int*)d_in[2];              // [N]
    int p = 3;
    const float *gatW[3], *gatAS[3], *gatAD[3], *gatB[3];
    for (int l = 0; l < 3; ++l) {
        gatW[l]  = (const float*)d_in[p++];
        gatAS[l] = (const float*)d_in[p++];
        gatAD[l] = (const float*)d_in[p++];
        gatB[l]  = (const float*)d_in[p++];
    }
    const float *projW[2], *projB[2], *projG[2], *projBe[2];
    for (int l = 0; l < 2; ++l) {
        projW[l]  = (const float*)d_in[p++];
        projB[l]  = (const float*)d_in[p++];
        projG[l]  = (const float*)d_in[p++];
        projBe[l] = (const float*)d_in[p++];
    }
    const float *mlpW[2], *mlpB[2], *mlpG[2], *mlpBe[2];
    for (int l = 0; l < 2; ++l) {
        mlpW[l]  = (const float*)d_in[p++];
        mlpB[l]  = (const float*)d_in[p++];
        mlpG[l]  = (const float*)d_in[p++];
        mlpBe[l] = (const float*)d_in[p++];
    }
    const float* outW = (const float*)d_in[p++];
    const float* outB = (const float*)d_in[p++];

    // -------- workspace carving (floats) --------
    float* w      = (float*)d_ws;
    float* Hlin   = w;                         // N*64  (reused as P1 after GAT layers)
    float* a_src  = Hlin  + (size_t)NN * HID;
    float* a_dst  = a_src + (size_t)NN * HEADS;
    float* mbuf   = a_dst + (size_t)NN * HEADS;
    float* denom  = mbuf  + (size_t)NN * HEADS;
    float* alpha  = denom + (size_t)NN * HEADS; // ETOT*4 (reused as P2 after GAT layers)
    float* Hcat   = alpha + (size_t)ETOT * HEADS;
    float* psum   = Hcat  + (size_t)NN * CATD;  // G*64
    float* pcnt   = psum  + GG * HID;           // G
    float* gbuf   = pcnt  + GG;                 // G*64
    float* gbuf2  = gbuf  + GG * HID;           // G*64
    float* logits = gbuf2 + GG * HID;           // G*40

    const int NH   = NN * HEADS;
    const int EH   = ETOT * HEADS;
    const int N64  = NN * HID;
    const dim3 gemmGrid((NN + 63) / 64);

    // zero the concat accumulator once (each layer owns a disjoint 64-col slot)
    fill_f32<<<gb(NN * CATD), 256, 0, stream>>>(Hcat, 0.f, NN * CATD);

    // -------- 3 GAT layers --------
    for (int l = 0; l < 3; ++l) {
        const float* Xl  = (l == 0) ? x : (Hcat + (l - 1) * HID);
        const int    ldx = (l == 0) ? INDIM : CATD;
        const int    K   = (l == 0) ? INDIM : HID;

        gemm_wmma_kernel<<<gemmGrid, 128, 0, stream>>>(
            Xl, ldx, gatW[l], nullptr, nullptr, nullptr, Hlin, HID, NN, K, 0);

        att_nodes_kernel<<<gb(NH), 256, 0, stream>>>(Hlin, gatAS[l], gatAD[l], a_src, a_dst);

        fill_f32<<<gb(NH), 256, 0, stream>>>(mbuf, -INFINITY, NH);
        fill_f32<<<gb(NH), 256, 0, stream>>>(denom, 0.f, NH);

        edge_alpha_max_kernel<<<gb(EH), 256, 0, stream>>>(e0, e1, a_src, a_dst, alpha, mbuf);
        edge_exp_sum_kernel  <<<gb(EH), 256, 0, stream>>>(e1, alpha, mbuf, denom);
        edge_aggregate_kernel<<<gb(EH), 256, 0, stream>>>(e0, e1, alpha, denom, Hlin, Hcat, l * HID);

        bias_elu_kernel<<<gb(N64), 256, 0, stream>>>(Hcat, gatB[l], l * HID);
    }

    // -------- projection MLP (fused bias+lrelu+BN epilogue, WMMA) --------
    float* P1 = Hlin;   // N*64, Hlin free now
    float* P2 = alpha;  // N*64 fits in ETOT*4 region
    gemm_wmma_kernel<<<gemmGrid, 128, 0, stream>>>(
        Hcat, CATD, projW[0], projB[0], projG[0], projBe[0], P1, HID, NN, CATD, 1);
    gemm_wmma_kernel<<<gemmGrid, 128, 0, stream>>>(
        P1, HID, projW[1], projB[1], projG[1], projBe[1], P2, HID, NN, HID, 1);

    // -------- global mean pool --------
    fill_f32<<<gb(GG * HID + GG), 256, 0, stream>>>(psum, 0.f, GG * HID + GG); // sums + cnt
    pool_sum_kernel<<<gb(N64), 256, 0, stream>>>(P2, batc, psum);
    pool_cnt_kernel<<<gb(NN), 256, 0, stream>>>(batc, pcnt);
    pool_div_kernel<<<gb(GG * HID), 256, 0, stream>>>(psum, pcnt, gbuf);

    // -------- head MLP + classifier + log softmax --------
    head_linear_kernel<<<gb(GG * HID), 256, 0, stream>>>(
        gbuf, mlpW[0], mlpB[0], mlpG[0], mlpBe[0], gbuf2, GG, HID, HID, 1);
    head_linear_kernel<<<gb(GG * HID), 256, 0, stream>>>(
        gbuf2, mlpW[1], mlpB[1], mlpG[1], mlpBe[1], gbuf, GG, HID, HID, 1);
    head_linear_kernel<<<gb(GG * OUTD), 256, 0, stream>>>(
        gbuf, outW, outB, nullptr, nullptr, logits, GG, HID, OUTD, 0);
    logsoftmax_kernel<<<GG, 64, 0, stream>>>(logits, (float*)d_out);
}
